// RelationalNetwork_63041529971195
// MI455X (gfx1250) — compile-verified, matrix operand order checked by reading, not run on hardware
//
#include <hip/hip_runtime.h>

// ---------------------------------------------------------------------------
// Relational Network on MI455X (gfx1250, wave32, WMMA bf16 + async-to-LDS)
//   B=32, N=64, D=256, Q=128, G=512, F=512, C=28
// Strategy:
//   pairs@W1 = x_i@W1a + x_j@W1b + q@W1c  -> precompute A_pre/B_pre/Cq
//   rel_sum  = sum_all_pairs relu(relu(h1)@W2+b2) - sum_diag(...)
//   main GEMM (69 GFLOP): v_wmma_f32_16x16x32_bf16, A from LDS h1 tile,
//   B (W2^T) streamed global->LDS via global_load_async_to_lds_b128 with
//   double buffering (ASYNCcnt), so WMMAs are fed by ds_load_b128.
// ---------------------------------------------------------------------------

typedef __bf16 bf16;
typedef __attribute__((ext_vector_type(4)))  bf16  v4bf;
typedef __attribute__((ext_vector_type(8)))  bf16  v8bf;
typedef __attribute__((ext_vector_type(16))) bf16  v16bf;
typedef __attribute__((ext_vector_type(8)))  float v8f;

#define B_ 32
#define N_ 64
#define D_ 256
#define Q_ 128
#define G_ 512
#define F_ 512
#define C_ 28

#define KCHUNK 32                    // K elements staged per buffer
#define WBUF_ELEMS (G_ * KCHUNK)     // 512 cols x 32 K = 32 KB bf16

// ---------------------------------------------------------------------------
// Fragment loader for 16-bit WMMA operands, row-major source (row = matrix
// row for A, row = output column for pre-transposed B).
// ISA 16-bit A 16x32 layout: lane L -> m = L&15, half = L>>4;
//   elems 0..7  = K (kc + half*8 .. +7)
//   elems 8..15 = K (kc + half*8 + 16 .. +23)
// ---------------------------------------------------------------------------
__device__ __forceinline__ v16bf load_frag(const bf16* __restrict__ base,
                                           int stride, int row, int kc, int lane) {
  const int kb = kc + ((lane >> 4) << 3);
  const bf16* p = base + (size_t)row * stride + kb;
  v8bf lo = *(const v8bf*)(p);
  v8bf hi = *(const v8bf*)(p + 16);
  return __builtin_shufflevector(lo, hi, 0, 1, 2, 3, 4, 5, 6, 7,
                                          8, 9, 10, 11, 12, 13, 14, 15);
}

// Async copy: one b128 per lane, global -> LDS, tracked by ASYNCcnt.
__device__ __forceinline__ void async_b128(unsigned lds_addr, const bf16* gaddr) {
  asm volatile("global_load_async_to_lds_b128 %0, %1, off"
               :: "v"(lds_addr), "v"(gaddr) : "memory");
}

// Stage one 512x32 bf16 chunk of W2t: 2048 b128 units, 4 per thread.
__device__ __forceinline__ void prefetch_w2(const bf16* __restrict__ W2t,
                                            bf16* wdst, int kc, int tid) {
#pragma unroll
  for (int k = 0; k < 4; ++k) {
    const int u16 = tid + k * 512;          // 0..2047
    const int n   = u16 >> 2;               // output column 0..511
    const int seg = u16 & 3;                // 8-elem segment within chunk
    const bf16* g = W2t + (size_t)n * G_ + kc + seg * 8;
    const unsigned lds = (unsigned)(uintptr_t)(wdst + n * KCHUNK + seg * 8);
    async_b128(lds, g);
  }
}

// ---------------------------------------------------------------------------
// Prep: bf16 conversions, weight transposes, rel_sum zeroing.
// ---------------------------------------------------------------------------
__global__ __launch_bounds__(256) void k_prep(const float* __restrict__ obj,
                                              const float* __restrict__ W1,
                                              const float* __restrict__ W2,
                                              bf16* __restrict__ Xbf,
                                              bf16* __restrict__ W1t,
                                              bf16* __restrict__ W2t,
                                              float* __restrict__ rel_sum) {
  const int idx = blockIdx.x * blockDim.x + threadIdx.x;
  if (idx < B_ * N_ * D_) Xbf[idx] = (bf16)obj[idx];
  if (idx < 1024 * 256) {
    const int n = idx >> 8, k = idx & 255;
    const float v = (n < G_) ? W1[k * G_ + n] : W1[(D_ + k) * G_ + (n - G_)];
    W1t[idx] = (bf16)v;
  }
  if (idx < G_ * G_) {
    const int n = idx >> 9, k = idx & 511;
    W2t[idx] = (bf16)W2[k * G_ + n];
  }
  if (idx < B_ * G_) rel_sum[idx] = 0.0f;
}

// Cq[b][g] = b_g1[g] + q[b] @ W1c   (tiny: 32x512)
__global__ __launch_bounds__(512) void k_cq(const float* __restrict__ qe,
                                            const float* __restrict__ W1,
                                            const float* __restrict__ b1,
                                            float* __restrict__ Cq) {
  const int b = blockIdx.x, g = threadIdx.x;
  float s = b1[g];
  for (int t = 0; t < Q_; ++t) s += qe[b * Q_ + t] * W1[(2 * D_ + t) * G_ + g];
  Cq[b * G_ + g] = s;
}

// ---------------------------------------------------------------------------
// Precompute GEMM:  AB_pre[2048 x 1024] = Xbf[2048 x 256] @ W1cat[256 x 1024]
// ---------------------------------------------------------------------------
__global__ __launch_bounds__(512) void k_pre_gemm(const bf16* __restrict__ Xbf,
                                                  const bf16* __restrict__ W1t,
                                                  float* __restrict__ AB_pre) {
  const int tid = threadIdx.x;
  const int wave = tid >> 5, lane = tid & 31;
  const int mb = blockIdx.x & 31;   // 32 blocks of 64 rows
  const int nb = blockIdx.x >> 5;   // 4 blocks of 256 cols
#pragma unroll
  for (int u = 0; u < 4; ++u) {
    const int t = wave * 4 + u;     // 64 wave-tiles
    const int mt = t >> 4;          // 0..3
    const int nt = t & 15;          // 0..15
    const int mrow = mb * 64 + mt * 16 + (lane & 15);
    const int nrow = nb * 256 + nt * 16 + (lane & 15);
    v8f acc = {0.f, 0.f, 0.f, 0.f, 0.f, 0.f, 0.f, 0.f};
#pragma unroll
    for (int kc = 0; kc < D_; kc += 32) {
      v16bf a = load_frag(Xbf, D_, mrow, kc, lane);
      v16bf w = load_frag(W1t, D_, nrow, kc, lane);
      acc = __builtin_amdgcn_wmma_f32_16x16x32_bf16(false, a, false, w,
                                                    (short)0, acc, false, false);
    }
    const int m0 = mb * 64 + mt * 16 + ((lane >> 4) << 3);
#pragma unroll
    for (int r = 0; r < 8; ++r)
      AB_pre[(size_t)(m0 + r) * 1024 + nrow] = acc[r];
  }
}

// ---------------------------------------------------------------------------
// Main fused kernel: one WG = one (b,i) with all 64 j's (64 pair-rows).
//   h1 tile (64 x 512 bf16, LDS) = relu(A_pre[i] + B_pre[j] + Cq)
//   W2t streamed global->LDS (async, double buffered, 16 chunks of K=32),
//   16 waves x (1 A-frag + 8 B-frags + 8 WMMA) per chunk,
//   fused bias+relu+row-sum epilogue, ds_add_f32 reduce, global atomic.
// diagMode=0: grid B*64, b=blk>>6, i=blk&63, j=r       (all pairs)
// diagMode=1: grid B,    b=blk,    i=j=r               (subtract diag)
// ---------------------------------------------------------------------------
__global__ __launch_bounds__(512) void k_rel_gemm(const float* __restrict__ AB_pre,
                                                  const float* __restrict__ Cq,
                                                  const bf16*  __restrict__ W2t,
                                                  const float* __restrict__ b2,
                                                  float* __restrict__ rel_sum,
                                                  int diagMode) {
  __shared__ __attribute__((aligned(16))) bf16  h1[64 * G_];          // 64 KB
  __shared__ __attribute__((aligned(16))) bf16  wbuf[2 * WBUF_ELEMS]; // 2x32 KB
  __shared__ float partial[G_];                                       //  2 KB

  const int tid = threadIdx.x;
  const int blk = blockIdx.x;
  const int b       = diagMode ? blk : (blk >> 6);
  const int i_fixed = diagMode ? 0   : (blk & 63);

  // kick off first W2 chunk while we build h1
  prefetch_w2(W2t, wbuf, 0, tid);

  if (tid < G_) partial[tid] = 0.0f;

  // ---- build h1 tile (vectorized x4): 64 rows x 512 cols ----
  for (int e4 = tid; e4 < (64 * G_) / 4; e4 += 512) {
    const int e = e4 * 4;
    const int r = e >> 9;           // 0..63  (= j)
    const int c = e & 511;
    const int i = diagMode ? r : i_fixed;
    const int j = r;
    const float4 av = *(const float4*)(AB_pre + (size_t)(b * N_ + i) * 1024 + c);
    const float4 bv = *(const float4*)(AB_pre + (size_t)(b * N_ + j) * 1024 + G_ + c);
    const float4 cv = *(const float4*)(Cq + b * G_ + c);
    v4bf o;
    o[0] = (bf16)fmaxf(av.x + bv.x + cv.x, 0.0f);
    o[1] = (bf16)fmaxf(av.y + bv.y + cv.y, 0.0f);
    o[2] = (bf16)fmaxf(av.z + bv.z + cv.z, 0.0f);
    o[3] = (bf16)fmaxf(av.w + bv.w + cv.w, 0.0f);
    *(v4bf*)(&h1[r * G_ + c]) = o;
  }

  // ---- WMMA GEMM over K=512 in 16 chunks, double-buffered W2 in LDS ----
  const int wave = tid >> 5, lane = tid & 31;
  const int mt = wave >> 2;          // 0..3 : fixed M-tile per wave
  const int nt0 = (wave & 3) * 8;    // 8 consecutive N-tiles per wave
  const int mrow = mt * 16 + (lane & 15);

  v8f acc[8];
#pragma unroll
  for (int u = 0; u < 8; ++u) acc[u] = (v8f){0.f, 0.f, 0.f, 0.f, 0.f, 0.f, 0.f, 0.f};

#pragma unroll 2
  for (int c = 0; c < G_ / KCHUNK; ++c) {
    if (c + 1 < G_ / KCHUNK) {
      prefetch_w2(W2t, wbuf + ((c + 1) & 1) * WBUF_ELEMS, (c + 1) * KCHUNK, tid);
      asm volatile("s_wait_asynccnt 0x4" ::: "memory");  // chunk c landed
    } else {
      asm volatile("s_wait_asynccnt 0x0" ::: "memory");
    }
    __syncthreads();                                     // visible to all waves

    const bf16* wb = wbuf + (c & 1) * WBUF_ELEMS;
    const v16bf a = load_frag(h1, G_, mrow, c * KCHUNK, lane);
#pragma unroll
    for (int u = 0; u < 8; ++u) {
      const v16bf w = load_frag(wb, KCHUNK, (nt0 + u) * 16 + (lane & 15), 0, lane);
      acc[u] = __builtin_amdgcn_wmma_f32_16x16x32_bf16(false, a, false, w,
                                                       (short)0, acc[u], false, false);
    }
    __syncthreads();                 // protect buffer from next overwrite
  }

  // ---- epilogue: bias + relu, reduce the 8 M-rows each lane holds ----
#pragma unroll
  for (int u = 0; u < 8; ++u) {
    const int ncol = (nt0 + u) * 16 + (lane & 15);
    const float bias = b2[ncol];
    float s = 0.0f;
#pragma unroll
    for (int r = 0; r < 8; ++r) s += fmaxf(acc[u][r] + bias, 0.0f);
    atomicAdd(&partial[ncol], s);    // ds_add_f32
  }
  __syncthreads();

  const float sign = diagMode ? -1.0f : 1.0f;
  if (tid < G_) atomicAdd(&rel_sum[b * G_ + tid], sign * partial[tid]);
}

// ---------------------------------------------------------------------------
// f_phi (tiny, fp32 VALU)
// ---------------------------------------------------------------------------
__global__ __launch_bounds__(512) void k_f1(const float* __restrict__ rel_sum,
                                            const float* __restrict__ Wf1,
                                            const float* __restrict__ bf1,
                                            float* __restrict__ f1) {
  const int b = blockIdx.x, n = threadIdx.x;
  float s = bf1[n];
  for (int k = 0; k < G_; ++k) s += rel_sum[b * G_ + k] * Wf1[k * F_ + n];
  f1[b * F_ + n] = fmaxf(s, 0.0f);
}

__global__ __launch_bounds__(32) void k_f2(const float* __restrict__ f1,
                                           const float* __restrict__ Wf2,
                                           const float* __restrict__ bf2,
                                           float* __restrict__ out) {
  const int b = blockIdx.x, c = threadIdx.x;
  if (c >= C_) return;
  float s = bf2[c];
  for (int k = 0; k < F_; ++k) s += f1[b * F_ + k] * Wf2[k * C_ + c];
  out[b * C_ + c] = s;
}

// ---------------------------------------------------------------------------
extern "C" void kernel_launch(void* const* d_in, const int* in_sizes, int n_in,
                              void* d_out, int out_size, void* d_ws, size_t ws_size,
                              hipStream_t stream) {
  const float* obj  = (const float*)d_in[0];
  const float* qe   = (const float*)d_in[1];
  const float* Wg1  = (const float*)d_in[2];
  const float* bg1  = (const float*)d_in[3];
  const float* Wg2  = (const float*)d_in[4];
  const float* bg2  = (const float*)d_in[5];
  const float* Wf1  = (const float*)d_in[6];
  const float* bf1v = (const float*)d_in[7];
  const float* Wf2  = (const float*)d_in[8];
  const float* bf2v = (const float*)d_in[9];
  float* out = (float*)d_out;

  char* ws = (char*)d_ws;
  size_t off = 0;
  auto alloc = [&](size_t bytes) -> void* {
    off = (off + 255) & ~(size_t)255;
    void* p = ws + off;
    off += bytes;
    return p;
  };
  bf16*  Xbf    = (bf16*) alloc((size_t)B_ * N_ * D_ * sizeof(bf16));    // 1 MB
  bf16*  W1t    = (bf16*) alloc((size_t)1024 * 256 * sizeof(bf16));      // 512 KB
  bf16*  W2t    = (bf16*) alloc((size_t)G_ * G_ * sizeof(bf16));         // 512 KB
  float* AB_pre = (float*)alloc((size_t)B_ * N_ * 1024 * sizeof(float)); // 8 MB
  float* Cq     = (float*)alloc((size_t)B_ * G_ * sizeof(float));
  float* rel    = (float*)alloc((size_t)B_ * G_ * sizeof(float));
  float* f1b    = (float*)alloc((size_t)B_ * F_ * sizeof(float));

  // 1) conversions / transposes / zero rel_sum
  k_prep<<<(B_ * N_ * D_) / 256, 256, 0, stream>>>(obj, Wg1, Wg2, Xbf, W1t, W2t, rel);
  // 2) Cq = b_g1 + q @ W1c
  k_cq<<<B_, G_, 0, stream>>>(qe, Wg1, bg1, Cq);
  // 3) AB_pre = X @ [W1a | W1b]   (WMMA)
  k_pre_gemm<<<128, 512, 0, stream>>>(Xbf, W1t, AB_pre);
  // 4) all-pairs fused layer-2 GEMM + reduction (WMMA, async-LDS W2 stream)
  k_rel_gemm<<<B_ * N_, 512, 0, stream>>>(AB_pre, Cq, W2t, bg2, rel, 0);
  // 5) subtract diagonal pairs
  k_rel_gemm<<<B_, 512, 0, stream>>>(AB_pre, Cq, W2t, bg2, rel, 1);
  // 6) f_phi
  k_f1<<<B_, F_, 0, stream>>>(rel, Wf1, bf1v, f1b);
  k_f2<<<B_, 32, 0, stream>>>(f1b, Wf2, bf2v, out);
}
